// Attention_10660108829265
// MI455X (gfx1250) — matrix operand order repaired and checked by reading, not compiled, not run on hardware
//
#include <hip/hip_runtime.h>

#define C_DIM 128
#define C2    256
#define S_DIM 4096
#define B_DIM 2
#define HD    32
#define TOK_PER_BLK 64
#define THREADS 128

typedef __attribute__((ext_vector_type(16))) _Float16 v16h;
typedef __attribute__((ext_vector_type(8)))  _Float16 v8h;
typedef __attribute__((ext_vector_type(8)))  float    v8f;

__device__ __forceinline__ v8f wmma_f16(v16h a, v16h b, v8f c) {
    // D = A(16x32) * B(32x16) + C, f32 accum
    return __builtin_amdgcn_wmma_f32_16x16x32_f16(false, a, false, b, (short)0, c, false, false);
}

__device__ __forceinline__ v16h cat16(v8h a, v8h b) {
    return __builtin_shufflevector(a, b, 0, 1, 2, 3, 4, 5, 6, 7, 8, 9, 10, 11, 12, 13, 14, 15);
}

// A-fragment (16x32 f16), source row-major [M][K]:
// lane L holds row M=L&15; elements e<8 -> k=hi*8+e, e>=8 -> k=16+hi*8+(e-8)
__device__ __forceinline__ v16h load_a(const _Float16* base, int rowstride) {
    const int lane = threadIdx.x & 31;
    const int row  = lane & 15;
    const int hi   = (lane >> 4) & 1;
    const _Float16* p = base + (size_t)row * rowstride + hi * 8;
    v8h a0 = *(const v8h*)p;
    v8h a1 = *(const v8h*)(p + 16);
    return cat16(a0, a1);
}

// B-fragment (32x16 f16): for column n, the 32 K-values are contiguous at
// base + n*rowstride; elements e<8 -> k=hi*8+e, e>=8 -> k=16+hi*8+(e-8).
__device__ __forceinline__ v16h load_b(const _Float16* base, int rowstride) {
    const int lane = threadIdx.x & 31;
    const int col  = lane & 15;
    const int hi   = (lane >> 4) & 1;
    const _Float16* p = base + (size_t)col * rowstride + hi * 8;
    v8h b0 = *(const v8h*)p;
    v8h b1 = *(const v8h*)(p + 16);
    return cat16(b0, b1);
}

// ---------------- Weight convert + transpose: f32 [K][N] -> f16 [N][K] -------------
__global__ void __launch_bounds__(256)
wcvt_kernel(const float* __restrict__ w, _Float16* __restrict__ wT, int K, int N) {
    int idx = blockIdx.x * 256 + threadIdx.x;
    if (idx < K * N) {
        int k = idx / N, n = idx - k * N;
        wT[(size_t)n * K + k] = (_Float16)w[idx];
    }
}

// ---------------- Projection: LN -> GEMM(128->256) -> LeakyReLU -> GEMM(256->128) ----
__global__ void __launch_bounds__(THREADS)
proj_kernel(const float* __restrict__ src,          // [B,C,S] (BCHW flat)
            const float* __restrict__ ln_g, const float* __restrict__ ln_b,
            const _Float16* __restrict__ w1T,       // f16 [2C][C]
            const float* __restrict__ b1,
            const _Float16* __restrict__ w2T,       // f16 [C][2C]
            const float* __restrict__ b2,
            _Float16* __restrict__ out16,           // nullable [B*S, C] f16 (token-major)
            _Float16* __restrict__ out16_bchw,      // nullable [B,C,S] f16
            float* __restrict__ out32)              // nullable [B*S, C] f32
{
    __shared__ alignas(16) _Float16 Xs[TOK_PER_BLK * C_DIM];    // 16 KB
    __shared__ alignas(16) _Float16 H1s[TOK_PER_BLK * C2];      // 32 KB
    float* Xraw = (float*)H1s;   // [C_DIM][TOK_PER_BLK] fp32 overlay; dead before H1s use
    const int tid  = threadIdx.x;
    const int tok0 = blockIdx.x * TOK_PER_BLK;
    const int blkb = tok0 >> 12;            // all 64 tokens share batch index
    const int s0   = tok0 & (S_DIM - 1);

    // coalesced float4 load of the [128 ch][64 tok] slab into LDS
    const float* slab = src + (size_t)blkb * C_DIM * S_DIM + s0;
    for (int idx = tid; idx < C_DIM * TOK_PER_BLK / 4; idx += THREADS) {
        int c  = idx >> 4;                  // 16 float4 per channel
        int t4 = idx & 15;
        ((float4*)Xraw)[idx] = *(const float4*)(slab + (size_t)c * S_DIM + t4 * 4);
    }
    __syncthreads();

    if (tid < TOK_PER_BLK) {
        float sum = 0.f, ssq = 0.f;
        for (int c = 0; c < C_DIM; ++c) {
            float v = Xraw[c * TOK_PER_BLK + tid];
            sum += v; ssq += v * v;
        }
        float mu   = sum * (1.f / C_DIM);
        float var  = ssq * (1.f / C_DIM) - mu * mu;
        float rstd = rsqrtf(var + 1e-5f);
        for (int c = 0; c < C_DIM; ++c) {
            float v = (Xraw[c * TOK_PER_BLK + tid] - mu) * rstd * ln_g[c] + ln_b[c];
            Xs[tid * C_DIM + c] = (_Float16)v;
        }
    }
    __syncthreads();    // Xs ready; Xraw (aliasing H1s) dead past this point

    const int wave = tid >> 5;
    const int lane = tid & 31;
    const int col  = lane & 15;
    const int hi   = (lane >> 4) & 1;
    const int m0   = wave * 16;

    // GEMM1: [16x128] x [128x256], LeakyReLU, stash f16 in LDS (per-wave band)
    for (int nb = 0; nb < 16; ++nb) {
        float bias = b1[nb * 16 + col];
        v8f acc;
#pragma unroll
        for (int r = 0; r < 8; ++r) acc[r] = bias;
#pragma unroll
        for (int kb = 0; kb < 4; ++kb) {
            v16h a  = load_a(&Xs[m0 * C_DIM + kb * 32], C_DIM);
            v16h bf = load_b(w1T + (size_t)(nb * 16) * C_DIM + kb * 32, C_DIM);
            acc = wmma_f16(a, bf, acc);
        }
#pragma unroll
        for (int r = 0; r < 8; ++r) {
            float v = acc[r];
            v = fmaxf(v, 0.01f * v);                // LeakyReLU(0.01)
            int m = r + hi * 8;
            H1s[(m0 + m) * C2 + nb * 16 + col] = (_Float16)v;
        }
    }
    __asm__ volatile("s_wait_dscnt 0" ::: "memory");

    // GEMM2: [16x256] x [256x128]
    for (int nb = 0; nb < 8; ++nb) {
        float bias = b2[nb * 16 + col];
        v8f acc;
#pragma unroll
        for (int r = 0; r < 8; ++r) acc[r] = bias;
#pragma unroll
        for (int kb = 0; kb < 8; ++kb) {
            v16h a  = load_a(&H1s[m0 * C2 + kb * 32], C2);
            v16h bf = load_b(w2T + (size_t)(nb * 16) * C2 + kb * 32, C2);
            acc = wmma_f16(a, bf, acc);
        }
#pragma unroll
        for (int r = 0; r < 8; ++r) {
            int m = r + hi * 8;
            int gtok = tok0 + m0 + m;
            int cc = nb * 16 + col;
            size_t idx = (size_t)gtok * C_DIM + cc;
            if (out16) out16[idx] = (_Float16)acc[r];
            if (out32) out32[idx] = acc[r];
            if (out16_bchw) {
                int ss = gtok & (S_DIM - 1);
                out16_bchw[((size_t)blkb * C_DIM + cc) * S_DIM + ss] = (_Float16)acc[r];
            }
        }
    }
}

// ---------------- Flash attention: 4 heads, d=32, online softmax -------------------
__global__ void __launch_bounds__(THREADS)
attn_kernel(const _Float16* __restrict__ qp,       // [B*S, C] token-major
            const _Float16* __restrict__ kp,       // [B*S, C] token-major
            const _Float16* __restrict__ vbchw,    // [B, C, S] (per-lane contiguous keys)
            _Float16* __restrict__ xo)             // [B*S, C] f16
{
    __shared__ alignas(16) _Float16 Ps[4 * 16 * 32];            // per-wave P tile, 4 KB
    const int tid  = threadIdx.x;
    const int wave = tid >> 5;
    const int lane = tid & 31;
    const int lo16 = lane & 15;
    const int hi   = (lane >> 4) & 1;

    const int qt = blockIdx.x & 63;
    const int bh = blockIdx.x >> 6;
    const int b  = bh >> 2;
    const int h  = bh & 3;
    const int q0 = qt * 64 + wave * 16;

    // Q fragment: A-layout, scaled by 1/sqrt(d) in f16 (values ~N(0,1), safe)
    v16h aq = load_a(qp + ((size_t)(b * S_DIM + q0)) * C_DIM + h * HD, C_DIM);
#pragma unroll
    for (int e = 0; e < 16; ++e) aq[e] = aq[e] * (_Float16)0.17677669529663689f;

    // ones B-fragment: osum = P @ 1 gives the softmax row-sum l_i via the XDL pipe
    v16h bones;
#pragma unroll
    for (int e = 0; e < 16; ++e) bones[e] = (_Float16)1.0f;

    v8f o0 = {}, o1 = {}, osum = {};
    float mi[8];
#pragma unroll
    for (int r = 0; r < 8; ++r) mi[r] = -__builtin_inff();

    _Float16* Pw = &Ps[wave * 16 * 32];
    const _Float16* vhead = vbchw + ((size_t)(b * C_DIM + h * HD)) * S_DIM;

    for (int j = 0; j < S_DIM / 32; ++j) {
        const int k0 = j * 32;
        const _Float16* kbase = kp + ((size_t)(b * S_DIM + k0)) * C_DIM + h * HD;

        // S = Q K^T : B cols = keys, 32 contiguous head-dims per lane
        v16h bk0 = load_b(kbase, C_DIM);
        v16h bk1 = load_b(kbase + 16 * C_DIM, C_DIM);
        v8f s0 = {}, s1 = {};
        s0 = wmma_f16(aq, bk0, s0);
        s1 = wmma_f16(aq, bk1, s1);

        // online softmax (row = C-layout: m = r + 8*hi, n = lane&15)
#pragma unroll
        for (int r = 0; r < 8; ++r) {
            float loc = fmaxf(s0[r], s1[r]);
            loc = fmaxf(loc, __shfl_xor(loc, 1, 16));
            loc = fmaxf(loc, __shfl_xor(loc, 2, 16));
            loc = fmaxf(loc, __shfl_xor(loc, 4, 16));
            loc = fmaxf(loc, __shfl_xor(loc, 8, 16));
            float mnew  = fmaxf(mi[r], loc);
            float alpha = __expf(mi[r] - mnew);
            mi[r] = mnew;
            float p0 = __expf(s0[r] - mnew);
            float p1 = __expf(s1[r] - mnew);
            o0[r]   *= alpha;
            o1[r]   *= alpha;
            osum[r] *= alpha;
            int m = r + hi * 8;
            Pw[m * 32 + lo16]      = (_Float16)p0;
            Pw[m * 32 + 16 + lo16] = (_Float16)p1;
        }
        __asm__ volatile("s_wait_dscnt 0" ::: "memory");

        // P (16x32, via LDS layout swap) @ [V | 1] (32x(32+16))
        v16h ap  = load_a(Pw, 32);
        v16h bv0 = load_b(vhead + k0, S_DIM);                      // dims 0..15
        v16h bv1 = load_b(vhead + (size_t)16 * S_DIM + k0, S_DIM); // dims 16..31
        o0   = wmma_f16(ap, bv0, o0);
        o1   = wmma_f16(ap, bv1, o1);
        osum = wmma_f16(ap, bones, osum);
    }

#pragma unroll
    for (int r = 0; r < 8; ++r) {
        float inv = 1.f / osum[r];
        int m = r + hi * 8;
        size_t base = ((size_t)(b * S_DIM + q0 + m)) * C_DIM + h * HD;
        xo[base + lo16]      = (_Float16)(o0[r] * inv);
        xo[base + 16 + lo16] = (_Float16)(o1[r] * inv);
    }
}

// ---------------- Residual MLP: out = res + Linear(LeakyReLU(Linear(x))) ------------
__global__ void __launch_bounds__(THREADS)
mlp_kernel(const _Float16* __restrict__ xin,        // [B*S, C] f16
           const float* __restrict__ res,           // [B*S, C] f32
           const _Float16* __restrict__ w1T, const float* __restrict__ b1,
           const _Float16* __restrict__ w2T, const float* __restrict__ b2,
           float* __restrict__ out32,               // nullable [B*S,C]
           _Float16* __restrict__ out16,            // nullable [B*S,C]
           float* __restrict__ out_bchw)            // nullable [B,C,S]
{
    __shared__ alignas(16) _Float16 Xs[TOK_PER_BLK * C_DIM];
    __shared__ alignas(16) _Float16 H1s[TOK_PER_BLK * C2];
    const int tid  = threadIdx.x;
    const int tok0 = blockIdx.x * TOK_PER_BLK;

    for (int i = tid; i < TOK_PER_BLK * C_DIM / 8; i += THREADS)
        ((v8h*)Xs)[i] = ((const v8h*)(xin + (size_t)tok0 * C_DIM))[i];
    __syncthreads();

    const int wave = tid >> 5;
    const int lane = tid & 31;
    const int col  = lane & 15;
    const int hi   = (lane >> 4) & 1;
    const int m0   = wave * 16;

    for (int nb = 0; nb < 16; ++nb) {
        float bias = b1[nb * 16 + col];
        v8f acc;
#pragma unroll
        for (int r = 0; r < 8; ++r) acc[r] = bias;
#pragma unroll
        for (int kb = 0; kb < 4; ++kb) {
            v16h a  = load_a(&Xs[m0 * C_DIM + kb * 32], C_DIM);
            v16h bf = load_b(w1T + (size_t)(nb * 16) * C_DIM + kb * 32, C_DIM);
            acc = wmma_f16(a, bf, acc);
        }
#pragma unroll
        for (int r = 0; r < 8; ++r) {
            float v = acc[r];
            v = fmaxf(v, 0.01f * v);
            int m = r + hi * 8;
            H1s[(m0 + m) * C2 + nb * 16 + col] = (_Float16)v;
        }
    }
    __asm__ volatile("s_wait_dscnt 0" ::: "memory");

    for (int nb = 0; nb < 8; ++nb) {
        float bias = b2[nb * 16 + col];
        v8f acc;
#pragma unroll
        for (int r = 0; r < 8; ++r) acc[r] = bias;
#pragma unroll
        for (int kb = 0; kb < 8; ++kb) {
            v16h a  = load_a(&H1s[m0 * C2 + kb * 32], C2);
            v16h bf = load_b(w2T + (size_t)(nb * 16) * C2 + kb * 32, C2);
            acc = wmma_f16(a, bf, acc);
        }
#pragma unroll
        for (int r = 0; r < 8; ++r) {
            int m = r + hi * 8;
            int gtok = tok0 + m0 + m;
            int cc = nb * 16 + col;
            size_t idx = (size_t)gtok * C_DIM + cc;
            float v = acc[r] + res[idx];
            if (out32) out32[idx] = v;
            if (out16) out16[idx] = (_Float16)v;
            if (out_bchw) {
                int bb = gtok >> 12;
                int ss = gtok & (S_DIM - 1);
                out_bchw[((size_t)bb * C_DIM + cc) * S_DIM + ss] = v;
            }
        }
    }
}

extern "C" void kernel_launch(void* const* d_in, const int* in_sizes, int n_in,
                              void* d_out, int out_size, void* d_ws, size_t ws_size,
                              hipStream_t stream) {
    (void)in_sizes; (void)n_in; (void)out_size; (void)ws_size;
    const float* q = (const float*)d_in[0];
    const float* k = (const float*)d_in[1];
    const float* v = (const float*)d_in[2];
    const float* ln_q_g = (const float*)d_in[3];
    const float* ln_q_b = (const float*)d_in[4];
    const float* wq1 = (const float*)d_in[5];
    const float* bq1 = (const float*)d_in[6];
    const float* wq2 = (const float*)d_in[7];
    const float* bq2 = (const float*)d_in[8];
    const float* ln_k_g = (const float*)d_in[9];
    const float* ln_k_b = (const float*)d_in[10];
    const float* wk1 = (const float*)d_in[11];
    const float* bk1 = (const float*)d_in[12];
    const float* wk2 = (const float*)d_in[13];
    const float* bk2 = (const float*)d_in[14];
    const float* ln_v_g = (const float*)d_in[15];
    const float* ln_v_b = (const float*)d_in[16];
    const float* wv1 = (const float*)d_in[17];
    const float* bv1 = (const float*)d_in[18];
    const float* wv2 = (const float*)d_in[19];
    const float* bv2 = (const float*)d_in[20];
    const float* m1_w1 = (const float*)d_in[21];
    const float* m1_b1 = (const float*)d_in[22];
    const float* m1_w2 = (const float*)d_in[23];
    const float* m1_b2 = (const float*)d_in[24];
    const float* m2_w1 = (const float*)d_in[25];
    const float* m2_b1 = (const float*)d_in[26];
    const float* m2_w2 = (const float*)d_in[27];
    const float* m2_b2 = (const float*)d_in[28];

    const size_t BS = (size_t)B_DIM * S_DIM;
    const size_t WELT = (size_t)C_DIM * C2;         // 32768 elements per weight
    char* ws = (char*)d_ws;
    _Float16* qp16   = (_Float16*)ws;  ws += BS * C_DIM * sizeof(_Float16);
    _Float16* kp16   = (_Float16*)ws;  ws += BS * C_DIM * sizeof(_Float16);
    _Float16* vbchw  = (_Float16*)ws;  ws += BS * C_DIM * sizeof(_Float16);
    float*    vp32   = (float*)ws;     ws += BS * C_DIM * sizeof(float);
    _Float16* x16    = (_Float16*)ws;  ws += BS * C_DIM * sizeof(_Float16);
    float*    rs132  = (float*)ws;     ws += BS * C_DIM * sizeof(float);
    _Float16* rs116  = (_Float16*)ws;  ws += BS * C_DIM * sizeof(_Float16);
    _Float16* wT[10];
    for (int i = 0; i < 10; ++i) { wT[i] = (_Float16*)ws; ws += WELT * sizeof(_Float16); }

    const int wblocks = (int)((WELT + 255) / 256);  // 128
    // transpose+convert: f32 [K][N] -> f16 [N][K]
    wcvt_kernel<<<wblocks, 256, 0, stream>>>(wq1,   wT[0], C_DIM, C2);
    wcvt_kernel<<<wblocks, 256, 0, stream>>>(wq2,   wT[1], C2, C_DIM);
    wcvt_kernel<<<wblocks, 256, 0, stream>>>(wk1,   wT[2], C_DIM, C2);
    wcvt_kernel<<<wblocks, 256, 0, stream>>>(wk2,   wT[3], C2, C_DIM);
    wcvt_kernel<<<wblocks, 256, 0, stream>>>(wv1,   wT[4], C_DIM, C2);
    wcvt_kernel<<<wblocks, 256, 0, stream>>>(wv2,   wT[5], C2, C_DIM);
    wcvt_kernel<<<wblocks, 256, 0, stream>>>(m1_w1, wT[6], C_DIM, C2);
    wcvt_kernel<<<wblocks, 256, 0, stream>>>(m1_w2, wT[7], C2, C_DIM);
    wcvt_kernel<<<wblocks, 256, 0, stream>>>(m2_w1, wT[8], C_DIM, C2);
    wcvt_kernel<<<wblocks, 256, 0, stream>>>(m2_w2, wT[9], C2, C_DIM);

    const int proj_blocks = (int)(BS / TOK_PER_BLK);            // 128
    const int attn_blocks = B_DIM * 4 * (S_DIM / TOK_PER_BLK);  // 512
    dim3 blk(THREADS);

    proj_kernel<<<proj_blocks, blk, 0, stream>>>(q, ln_q_g, ln_q_b, wT[0], bq1, wT[1], bq2,
                                                 qp16, nullptr, nullptr);
    proj_kernel<<<proj_blocks, blk, 0, stream>>>(k, ln_k_g, ln_k_b, wT[2], bk1, wT[3], bk2,
                                                 kp16, nullptr, nullptr);
    proj_kernel<<<proj_blocks, blk, 0, stream>>>(v, ln_v_g, ln_v_b, wT[4], bv1, wT[5], bv2,
                                                 nullptr, vbchw, vp32);
    attn_kernel<<<attn_blocks, blk, 0, stream>>>(qp16, kp16, vbchw, x16);
    mlp_kernel<<<proj_blocks, blk, 0, stream>>>(x16, vp32, wT[6], m1_b1, wT[7], m1_b2,
                                                rs132, rs116, nullptr);
    mlp_kernel<<<proj_blocks, blk, 0, stream>>>(rs116, rs132, wT[8], m2_b1, wT[9], m2_b2,
                                                nullptr, nullptr, (float*)d_out);
}
